// Decoder_56401510531330
// MI455X (gfx1250) — compile-verified, matrix operand order checked by reading, not run on hardware
//
#include <hip/hip_runtime.h>
#include <hip/hip_bf16.h>

// Problem dims
#define Bdim 32
#define Tdim 64
#define Sdim 64
#define Vdim 32000
#define Edim 512
#define Hdim 512
#define G4H  (4*Hdim)      // 2048
#define NEGV (-1e9f)

typedef __attribute__((ext_vector_type(16))) __bf16 v16bf;
typedef __attribute__((ext_vector_type(8)))  __bf16 v8bf;
typedef __attribute__((ext_vector_type(8)))  float  v8f;
typedef __attribute__((ext_vector_type(4)))  unsigned int u32x4;
typedef __attribute__((ext_vector_type(8)))  unsigned int u32x8;

// ---------------------------------------------------------------------------
// WMMA helpers (wave32). 16x16x32 bf16 -> f32 accumulate.
// A fragment (16x32, row-major source, K contiguous): lane = 16*hf + m,
//   elements = two contiguous 8-element (16B) chunks at K = 8*hf and 16+8*hf.
// B fragment built from row-major W[N,K] with identical addressing since
//   B[k][n] = W[n][k].   C/D: VGPR r, lane L: M = r + 8*(L/16), N = L%16.
// ---------------------------------------------------------------------------
__device__ __forceinline__ v16bf load_frag_rm(const __bf16* base, int row, int ld,
                                              int k0, int hf) {
    const __bf16* p = base + (size_t)row * ld + k0 + 8 * hf;
    v8bf lo = *(const v8bf*)(p);
    v8bf hi = *(const v8bf*)(p + 16);
    return __builtin_shufflevector(lo, hi, 0,1,2,3,4,5,6,7,8,9,10,11,12,13,14,15);
}

__device__ __forceinline__ void wmma_kloop(v8f& acc,
                                           const __bf16* A, int lda, int arow,
                                           const __bf16* W, int ldw, int wrow, int K) {
    const int lane = threadIdx.x & 31;
    const int l16  = lane & 15;
    const int hf   = lane >> 4;
    for (int k = 0; k < K; k += 32) {
        v16bf a = load_frag_rm(A, arow + l16, lda, k, hf);
        v16bf b = load_frag_rm(W, wrow + l16, ldw, k, hf);
        acc = __builtin_amdgcn_wmma_f32_16x16x32_bf16(
                  false, a, false, b, (short)0, acc, false, false);
    }
}

// ---------------------------------------------------------------------------
// Tensor Data Mover: 2D tile DMA (global -> LDS), D# per ISA ch.8.
// group0: [1:0]=count(1), [63:32]=lds_addr, [120:64]=global_addr, [127:126]=2
// group1: [17:16]=data_size(1 => 2B), [79:48]=tensor_dim0, [111:80]=tensor_dim1,
//         [127:112]=tile_dim0, [143:128]=tile_dim1, [207:160]=tensor_dim0_stride
// Groups 2/3 omitted (NULL) -> 2D tensor form.
// ---------------------------------------------------------------------------
__device__ __forceinline__ void tdm_load_2d(unsigned lds_off, const void* gptr,
                                            unsigned tile_d0, unsigned tile_d1,
                                            unsigned tens_d0, unsigned tens_d1,
                                            unsigned stride0) {
    unsigned long long ga = (unsigned long long)gptr;
    u32x4 g0;
    g0[0] = 1u;                                   // count=1, user mode
    g0[1] = lds_off;                              // LDS byte address
    g0[2] = (unsigned)ga;                         // global addr [31:0]
    g0[3] = (unsigned)((ga >> 32) & 0x01FFFFFFu) | 0x80000000u;  // addr[56:32] | type=2
    u32x8 g1;
    g1[0] = 1u << 16;                             // data_size=1 (2 bytes); no flags
    g1[1] = (tens_d0 & 0xFFFFu) << 16;            // tensor_dim0[15:0]
    g1[2] = (tens_d0 >> 16) | ((tens_d1 & 0xFFFFu) << 16);
    g1[3] = ((tens_d1 >> 16) & 0xFFFFu) | (tile_d0 << 16);
    g1[4] = tile_d1 & 0xFFFFu;                    // tile_dim1 ; tile_dim2=0
    g1[5] = stride0;                              // tensor_dim0_stride[31:0]
    g1[6] = 0u;                                   // stride0 hi | tensor_dim1_stride lo
    g1[7] = 0u;
    asm volatile("tensor_load_to_lds %0, %1" :: "s"(g0), "s"(g1) : "memory");
}

// ---------------------------------------------------------------------------
// Small kernels
// ---------------------------------------------------------------------------
__global__ void f32_to_bf16(const float* __restrict__ src, __bf16* __restrict__ dst, int n) {
    int i = blockIdx.x * blockDim.x + threadIdx.x;
    if (i < n) dst[i] = (__bf16)src[i];
}

__global__ void embed_gather(const int* __restrict__ tgt, const float* __restrict__ emb,
                             __bf16* __restrict__ x_bf) {
    int b = blockIdx.x / Tdim, t = blockIdx.x % Tdim;
    int tok = tgt[b * Tdim + t];
    const float* src = emb + (size_t)tok * Edim;
    __bf16* dst = x_bf + ((size_t)t * Bdim + b) * Edim;   // [T,B,E]
    for (int e = threadIdx.x; e < Edim; e += blockDim.x)
        dst[e] = (__bf16)src[e];
}

__global__ void init_state(const float* __restrict__ h0, const float* __restrict__ c0,
                           float* h_st, float* c_st, __bf16* h_bf, __bf16* oprev_bf) {
    int i = blockIdx.x * blockDim.x + threadIdx.x;
    if (i < 2 * Bdim * Hdim) {
        h_st[i] = h0[i];
        c_st[i] = c0[i];
        h_bf[i] = (__bf16)h0[i];
    }
    if (i < Bdim * Hdim) oprev_bf[i] = (__bf16)0.0f;
}

// gates[b,n] = A1@W1^T + A2@W2^T (+ A3@W3^T) + bias1[n] + bias2[n]
__global__ void gemm_gates(const __bf16* __restrict__ A1, const __bf16* __restrict__ W1, int ldw1, int K1,
                           const __bf16* __restrict__ A2, const __bf16* __restrict__ W2, int ldw2, int K2,
                           const __bf16* __restrict__ A3, const __bf16* __restrict__ W3, int ldw3, int K3,
                           const float* __restrict__ bias1, const float* __restrict__ bias2,
                           float* __restrict__ out, int N) {
    const int wave = threadIdx.x >> 5;
    const int tile = blockIdx.x * (blockDim.x >> 5) + wave;
    const int mt = tile & 1;           // 32 rows -> 2 M-tiles
    const int nt = tile >> 1;
    const int m0 = mt * 16, n0 = nt * 16;

    v8f acc = {};
    wmma_kloop(acc, A1, 512, m0, W1, ldw1, n0, K1);
    wmma_kloop(acc, A2, 512, m0, W2, ldw2, n0, K2);
    if (A3) wmma_kloop(acc, A3, 512, m0, W3, ldw3, n0, K3);

    const int lane = threadIdx.x & 31;
    const int l16 = lane & 15, hf = lane >> 4;
    const int col = n0 + l16;
    const float bsum = bias1[col] + bias2[col];
#pragma unroll
    for (int r = 0; r < 8; ++r) {
        int row = m0 + r + 8 * hf;
        out[row * N + col] = acc[r] + bsum;
    }
}

__global__ void lstm_elem(const float* __restrict__ gates, float* h_st, float* c_st,
                          __bf16* h_bf, __bf16* cat_dst /* nullable */) {
    int i = blockIdx.x * blockDim.x + threadIdx.x;   // b*H + h
    if (i >= Bdim * Hdim) return;
    int b = i / Hdim, h = i - b * Hdim;
    const float* g = gates + (size_t)b * G4H;
    float ig = 1.0f / (1.0f + __expf(-g[h]));
    float fg = 1.0f / (1.0f + __expf(-g[Hdim + h]));
    float gg = tanhf(g[2 * Hdim + h]);
    float og = 1.0f / (1.0f + __expf(-g[3 * Hdim + h]));
    float c  = fg * c_st[i] + ig * gg;
    float hn = og * tanhf(c);
    c_st[i] = c;
    h_st[i] = hn;
    h_bf[i] = (__bf16)hn;
    if (cat_dst) cat_dst[b * 1024 + h] = (__bf16)hn;
}

__global__ void attention_ctx(const float* __restrict__ enc, const unsigned char* __restrict__ mask,
                              const float* __restrict__ h1, float* __restrict__ attn_base, int t,
                              __bf16* __restrict__ cat_bf) {
    __shared__ float sc[Sdim];
    __shared__ float sa[Sdim];
    const int b = blockIdx.x;
    const int s = threadIdx.x;            // 0..63
    const float* er = enc + ((size_t)b * Sdim + s) * Hdim;
    const float* hr = h1 + (size_t)b * Hdim;
    float dot = 0.0f;
    for (int h = 0; h < Hdim; h += 4)
        dot += er[h] * hr[h] + er[h + 1] * hr[h + 1]
             + er[h + 2] * hr[h + 2] + er[h + 3] * hr[h + 3];
    if (!mask[b * Sdim + s]) dot = NEGV;
    sc[s] = dot;
    __syncthreads();
    float mx = -3.4e38f;
    for (int j = 0; j < Sdim; ++j) mx = fmaxf(mx, sc[j]);
    float e = __expf(dot - mx);
    sa[s] = e;
    __syncthreads();
    float sum = 0.0f;
    for (int j = 0; j < Sdim; ++j) sum += sa[j];
    float a = e / sum;
    attn_base[((size_t)b * Tdim + t) * Sdim + s] = a;
    __syncthreads();
    sa[s] = a;
    __syncthreads();
#pragma unroll
    for (int hh = 0; hh < 8; ++hh) {
        int h = s * 8 + hh;
        float cx = 0.0f;
        for (int j = 0; j < Sdim; ++j)
            cx += sa[j] * enc[((size_t)b * Sdim + j) * Hdim + h];
        cat_bf[b * 1024 + 512 + h] = (__bf16)cx;   // cat[:,512:1024]
    }
}

__global__ void gemm_o_tanh(const __bf16* __restrict__ Acat, const __bf16* __restrict__ Wg,
                            const float* __restrict__ bg,
                            __bf16* __restrict__ oprev_bf, __bf16* __restrict__ oseq_t) {
    const int wave = threadIdx.x >> 5;
    const int tile = blockIdx.x * (blockDim.x >> 5) + wave;   // 64 tiles
    const int mt = tile & 1, nt = tile >> 1;
    const int m0 = mt * 16, n0 = nt * 16;
    v8f acc = {};
    wmma_kloop(acc, Acat, 1024, m0, Wg, 1024, n0, 1024);
    const int lane = threadIdx.x & 31;
    const int l16 = lane & 15, hf = lane >> 4;
    const int col = n0 + l16;
    const float bb = bg[col];
#pragma unroll
    for (int r = 0; r < 8; ++r) {
        int row = m0 + r + 8 * hf;
        __bf16 bv = (__bf16)tanhf(acc[r] + bb);
        oprev_bf[row * Hdim + col] = bv;
        oseq_t[row * Hdim + col]   = bv;
    }
}

// ---------------------------------------------------------------------------
// Vocab projection: [2048 x 512] x [512 x 32000], LDS-tiled 64x128, K-chunk 64,
// double-buffered via TDM (tensor_load_to_lds + s_wait_tensorcnt).
// 8 waves/block: wave w -> M-subtile (w&3), N-half (w>>2) covering 4 N-subtiles.
// ---------------------------------------------------------------------------
__global__ void gemm_vocab_tdm(const __bf16* __restrict__ Oseq, const __bf16* __restrict__ Wv,
                               float* __restrict__ out) {
    __shared__ __bf16 bufA[2][64 * 64];     // 64 M-rows x 64 K
    __shared__ __bf16 bufB[2][128 * 64];    // 128 N-rows x 64 K
    const int wv   = threadIdx.x >> 5;
    const int lane = threadIdx.x & 31;
    const int l16  = lane & 15, hf = lane >> 4;
    const int m_blk = blockIdx.y * 64;
    const int n_blk = blockIdx.x * 128;
    const int NCH = Hdim / 64;              // 8 K-chunks

    v8f acc[4] = {v8f{}, v8f{}, v8f{}, v8f{}};

    if (wv == 0) {   // wave-uniform branch: wave 0 drives the TDM
        tdm_load_2d((unsigned)(size_t)&bufA[0][0], Oseq + (size_t)m_blk * Hdim,
                    64, 64, Hdim, Tdim * Bdim, Hdim);
        tdm_load_2d((unsigned)(size_t)&bufB[0][0], Wv + (size_t)n_blk * Hdim,
                    64, 128, Hdim, Vdim, Hdim);
    }

    for (int c = 0; c < NCH; ++c) {
        const int cur = c & 1;
        if (wv == 0) {
            if (c + 1 < NCH) {
                const int nxt = cur ^ 1;
                tdm_load_2d((unsigned)(size_t)&bufA[nxt][0],
                            Oseq + (size_t)m_blk * Hdim + (c + 1) * 64,
                            64, 64, Hdim, Tdim * Bdim, Hdim);
                tdm_load_2d((unsigned)(size_t)&bufB[nxt][0],
                            Wv + (size_t)n_blk * Hdim + (c + 1) * 64,
                            64, 128, Hdim, Vdim, Hdim);
                __builtin_amdgcn_s_wait_tensorcnt(2);  // previous pair landed (in-order)
            } else {
                __builtin_amdgcn_s_wait_tensorcnt(0);
            }
        }
        __syncthreads();                     // publish LDS tiles to all waves
        const __bf16* tA = &bufA[cur][0];
        const __bf16* tB = &bufB[cur][0];
#pragma unroll
        for (int kk = 0; kk < 64; kk += 32) {
            v16bf a = load_frag_rm(tA, 16 * (wv & 3) + l16, 64, kk, hf);
#pragma unroll
            for (int j = 0; j < 4; ++j) {
                v16bf b = load_frag_rm(tB, 64 * (wv >> 2) + 16 * j + l16, 64, kk, hf);
                acc[j] = __builtin_amdgcn_wmma_f32_16x16x32_bf16(
                             false, a, false, b, (short)0, acc[j], false, false);
            }
        }
        __syncthreads();                     // don't let next DMA overwrite live buffer
    }

#pragma unroll
    for (int j = 0; j < 4; ++j) {
        const int col = n_blk + 64 * (wv >> 2) + 16 * j + l16;
#pragma unroll
        for (int r = 0; r < 8; ++r) {
            int row = m_blk + 16 * (wv & 3) + r + 8 * hf;   // row = t*B + b
            int t = row >> 5, b = row & 31;                 // B == 32
            out[((size_t)b * Tdim + t) * Vdim + col] = acc[j][r];
        }
    }
}

__global__ void copy_fin(const float* __restrict__ h_st, const float* __restrict__ c_st,
                         float* __restrict__ dst_h, float* __restrict__ dst_c) {
    int i = blockIdx.x * blockDim.x + threadIdx.x;
    if (i < 2 * Bdim * Hdim) { dst_h[i] = h_st[i]; dst_c[i] = c_st[i]; }
}

// ---------------------------------------------------------------------------
// Host launcher
// ---------------------------------------------------------------------------
extern "C" void kernel_launch(void* const* d_in, const int* in_sizes, int n_in,
                              void* d_out, int out_size, void* d_ws, size_t ws_size,
                              hipStream_t stream) {
    const int*   tgt   = (const int*)  d_in[0];
    const float* h0    = (const float*)d_in[1];
    const float* c0    = (const float*)d_in[2];
    const float* enc   = (const float*)d_in[3];
    const unsigned char* mask = (const unsigned char*)d_in[4];
    const float* emb   = (const float*)d_in[5];
    const float* W_ih0 = (const float*)d_in[6];
    const float* W_hh0 = (const float*)d_in[7];
    const float* b_ih0 = (const float*)d_in[8];
    const float* b_hh0 = (const float*)d_in[9];
    const float* W_ih1 = (const float*)d_in[10];
    const float* W_hh1 = (const float*)d_in[11];
    const float* b_ih1 = (const float*)d_in[12];
    const float* b_hh1 = (const float*)d_in[13];
    const float* W_g   = (const float*)d_in[14];
    const float* b_g   = (const float*)d_in[15];
    const float* W_v   = (const float*)d_in[16];

    char* ws = (char*)d_ws;
    size_t off = 0;
    auto alloc = [&](size_t bytes) -> void* {
        void* p = ws + off;
        off += (bytes + 255) & ~(size_t)255;
        return p;
    };
    __bf16* Wih0_bf = (__bf16*)alloc((size_t)G4H * 1024 * 2);
    __bf16* Whh0_bf = (__bf16*)alloc((size_t)G4H * 512  * 2);
    __bf16* Wih1_bf = (__bf16*)alloc((size_t)G4H * 512  * 2);
    __bf16* Whh1_bf = (__bf16*)alloc((size_t)G4H * 512  * 2);
    __bf16* Wg_bf   = (__bf16*)alloc((size_t)Hdim * 1024 * 2);
    __bf16* Wv_bf   = (__bf16*)alloc((size_t)Vdim * Hdim * 2);
    __bf16* x_bf    = (__bf16*)alloc((size_t)Tdim * Bdim * Edim * 2);
    __bf16* oseq_bf = (__bf16*)alloc((size_t)Tdim * Bdim * Hdim * 2);
    float*  h_state = (float*) alloc((size_t)2 * Bdim * Hdim * 4);
    float*  c_state = (float*) alloc((size_t)2 * Bdim * Hdim * 4);
    __bf16* h_bf    = (__bf16*)alloc((size_t)2 * Bdim * Hdim * 2);
    __bf16* oprev_bf= (__bf16*)alloc((size_t)Bdim * Hdim * 2);
    __bf16* cat_bf  = (__bf16*)alloc((size_t)Bdim * 1024 * 2);
    float*  gates0  = (float*) alloc((size_t)Bdim * G4H * 4);
    float*  gates1  = (float*) alloc((size_t)Bdim * G4H * 4);

    // Output layout: outputs | h_fin | c_fin | attn
    float* out      = (float*)d_out;
    float* out_hfin = out + (size_t)Bdim * Tdim * Vdim;
    float* out_cfin = out_hfin + 2 * Bdim * Hdim;
    float* out_attn = out_cfin + 2 * Bdim * Hdim;

    f32_to_bf16<<<(G4H*1024 + 255)/256, 256, 0, stream>>>(W_ih0, Wih0_bf, G4H*1024);
    f32_to_bf16<<<(G4H*512  + 255)/256, 256, 0, stream>>>(W_hh0, Whh0_bf, G4H*512);
    f32_to_bf16<<<(G4H*512  + 255)/256, 256, 0, stream>>>(W_ih1, Wih1_bf, G4H*512);
    f32_to_bf16<<<(G4H*512  + 255)/256, 256, 0, stream>>>(W_hh1, Whh1_bf, G4H*512);
    f32_to_bf16<<<(Hdim*1024 + 255)/256, 256, 0, stream>>>(W_g, Wg_bf, Hdim*1024);
    f32_to_bf16<<<(Vdim*Hdim + 255)/256, 256, 0, stream>>>(W_v, Wv_bf, Vdim*Hdim);

    embed_gather<<<Bdim * Tdim, 128, 0, stream>>>(tgt, emb, x_bf);
    init_state<<<(2*Bdim*Hdim + 255)/256, 256, 0, stream>>>(h0, c0, h_state, c_state,
                                                            h_bf, oprev_bf);

    for (int t = 0; t < Tdim; ++t) {
        gemm_gates<<<64, 128, 0, stream>>>(
            x_bf + (size_t)t * Bdim * Edim, Wih0_bf,       1024, 512,
            oprev_bf,                        Wih0_bf + 512, 1024, 512,
            h_bf,                            Whh0_bf,       512,  512,
            b_ih0, b_hh0, gates0, G4H);
        lstm_elem<<<(Bdim*Hdim + 255)/256, 256, 0, stream>>>(
            gates0, h_state, c_state, h_bf, nullptr);
        gemm_gates<<<64, 128, 0, stream>>>(
            h_bf,                Wih1_bf, 512, 512,
            h_bf + Bdim*Hdim,    Whh1_bf, 512, 512,
            nullptr, nullptr, 512, 0,
            b_ih1, b_hh1, gates1, G4H);
        lstm_elem<<<(Bdim*Hdim + 255)/256, 256, 0, stream>>>(
            gates1, h_state + Bdim*Hdim, c_state + Bdim*Hdim, h_bf + Bdim*Hdim, cat_bf);
        attention_ctx<<<Bdim, Sdim, 0, stream>>>(
            enc, mask, h_state + Bdim*Hdim, out_attn, t, cat_bf);
        gemm_o_tanh<<<16, 128, 0, stream>>>(
            cat_bf, Wg_bf, b_g, oprev_bf, oseq_bf + (size_t)t * Bdim * Hdim);
    }

    // Vocab projection: LDS-tiled + TDM double-buffered WMMA GEMM
    gemm_vocab_tdm<<<dim3(Vdim/128, (Tdim*Bdim)/64), 256, 0, stream>>>(oseq_bf, Wv_bf, out);
    copy_fin<<<(2*Bdim*Hdim + 255)/256, 256, 0, stream>>>(h_state, c_state, out_hfin, out_cfin);
}
